// BiLSTM_CRF_8727373545639
// MI455X (gfx1250) — matrix operand order repaired and checked by reading, not compiled, not run on hardware
//
#include <hip/hip_runtime.h>
#include <hip/hip_bf16.h>

// Problem dims
#define LSEQ 512
#define BB   64
#define EE   512
#define HH   512
#define GG   2048   // 4*H
#define HD2  1024   // 2*H
#define TT   16
#define NEG_INF   -10000.0f
#define START_TAG 14
#define STOP_TAG  15

typedef __bf16 bf16_t;
typedef __attribute__((ext_vector_type(8)))  bf16_t v8bf;
typedef __attribute__((ext_vector_type(16))) bf16_t v16bf;
typedef __attribute__((ext_vector_type(8)))  float  v8f;

__device__ __forceinline__ float sigm(float x) { return 1.0f / (1.0f + __expf(-x)); }

__device__ __forceinline__ v16bf load16(const bf16_t* p) {
    v8bf lo = *(const v8bf*)p;          // K = kb .. kb+7
    v8bf hi = *(const v8bf*)(p + 16);   // K = kb+16 .. kb+23
    return __builtin_shufflevector(lo, hi, 0,1,2,3,4,5,6,7,8,9,10,11,12,13,14,15);
}
__device__ __forceinline__ v16bf load8x2(const bf16_t* p) {
    v8bf lo = *(const v8bf*)p;          // K = kb .. kb+7
    v8bf hi = *(const v8bf*)(p + 16);   // K = kb+16 .. kb+23
    return __builtin_shufflevector(lo, hi, 0,1,2,3,4,5,6,7,8,9,10,11,12,13,14,15);
}

// ---------------------------------------------------------------------------
// Kernel 0: embedding gather -> bf16, weight concat [Wih|Whh] -> bf16,
//           h0/c0 init, W_tag -> bf16.
// ---------------------------------------------------------------------------
__global__ __launch_bounds__(256)
void prep_kernel(const int* __restrict__ sentences, const float* __restrict__ embed,
                 const float* __restrict__ Wih_f, const float* __restrict__ Whh_f,
                 const float* __restrict__ Wih_b, const float* __restrict__ Whh_b,
                 const float* __restrict__ h0,    const float* __restrict__ c0,
                 const float* __restrict__ W_tag,
                 bf16_t* __restrict__ Xbf,   // [L][B][E]
                 bf16_t* __restrict__ Wcat,  // [2][G][2H]  (k<512: Wih, else Whh)
                 bf16_t* __restrict__ Hbuf,  // [parity=2][2][B][H] ; init parity 0
                 float*  __restrict__ Cbuf,  // [2][B][H]
                 bf16_t* __restrict__ Wtag)  // [T][HD2]
{
    const size_t stride = (size_t)gridDim.x * blockDim.x;
    const size_t i0 = (size_t)blockIdx.x * blockDim.x + threadIdx.x;

    // Embedding gather (X stored [t][b][e])
    for (size_t i = i0; i < (size_t)LSEQ * BB * EE; i += stride) {
        size_t e = i & (EE - 1);
        size_t tb = i >> 9;            // /EE
        size_t b = tb & (BB - 1);
        size_t t = tb >> 6;            // /BB
        int tok = sentences[b * LSEQ + t];
        Xbf[i] = (bf16_t)embed[(size_t)tok * EE + e];
    }
    // Concatenated recurrent weights, both directions
    for (size_t i = i0; i < (size_t)GG * HD2; i += stride) {
        size_t n = i >> 10;            // /HD2
        size_t k = i & (HD2 - 1);
        float vf = (k < EE) ? Wih_f[n * EE + k] : Whh_f[n * HH + (k - EE)];
        float vb = (k < EE) ? Wih_b[n * EE + k] : Whh_b[n * HH + (k - EE)];
        Wcat[i] = (bf16_t)vf;
        Wcat[(size_t)GG * HD2 + i] = (bf16_t)vb;
    }
    // h0 / c0 (parity-0 H buffer)
    for (size_t i = i0; i < (size_t)2 * BB * HH; i += stride) {
        Hbuf[i] = (bf16_t)h0[i];
        Cbuf[i] = c0[i];
    }
    // Tag weights
    for (size_t i = i0; i < (size_t)TT * HD2; i += stride)
        Wtag[i] = (bf16_t)W_tag[i];
}

// ---------------------------------------------------------------------------
// Kernel 1 (per step, FUSED): gate GEMM + LSTM cell.
//   Grid: 64 blocks x 512 threads. Block = (dir, j-slice of 16 hidden cols).
//   16 waves/block: wave w -> M-tile (w&3), gate chunk (w>>2).
//   Gates staged in LDS (never touch global); cell update done in-block.
//   H double-buffered by step parity (Hread / Hwrite).
// ---------------------------------------------------------------------------
__global__ __launch_bounds__(512)
void lstm_step_kernel(const bf16_t* __restrict__ Xbf,
                      const bf16_t* __restrict__ Wcat,
                      const bf16_t* __restrict__ Hread,   // [2][B][H] bf16
                      bf16_t* __restrict__ Hwrite,        // [2][B][H] bf16
                      float*  __restrict__ Cbuf,          // [2][B][H] f32
                      bf16_t* __restrict__ hs,            // [L][B][HD2]
                      const float* __restrict__ bias_f,
                      const float* __restrict__ bias_b,
                      int s)
{
    __shared__ float gsm[4][BB][17];   // [gate][m][j] padded vs bank conflicts

    const int tid  = threadIdx.x;
    const int lane = tid & 31;
    const int wave = tid >> 5;          // 0..15
    const int mt   = wave & 3;          // M tile (rows mt*16..+15)
    const int gc   = wave >> 2;         // gate chunk: 0=i,1=f,2=g,3=o
    const int dir  = blockIdx.x >> 5;   // 0=fwd, 1=bwd
    const int jblk = blockIdx.x & 31;   // 16-wide hidden-column slice
    const int t    = dir ? (LSEQ - 1 - s) : s;

    const int l15 = lane & 15;
    const int ka  = (lane >> 4) * 8;    // A sub-offset in 32-wide K chunk
    const int kb  = (lane >> 4) * 8;    // B sub-offset
    const int nrow = gc * HH + jblk * 16 + l15;   // gate output column (0..2047)

    const bf16_t* xrow = Xbf   + ((size_t)t * BB + mt * 16 + l15) * EE;
    const bf16_t* hrow = Hread + ((size_t)dir * BB + mt * 16 + l15) * HH;
    const bf16_t* wrow = Wcat  + ((size_t)dir * GG + nrow) * HD2;

    // Prefetch next step's activation rows (only gc==0 waves; 1KB/row; upper lane-half
    // prefetches at +1KB). Lowers to global_prefetch_b8.
    const int tn = dir ? (t - 1) : (t + 1);
    if (gc == 0 && tn >= 0 && tn < LSEQ) {
        const char* pn = (const char*)(Xbf + ((size_t)tn * BB + mt * 16 + l15) * EE);
        __builtin_prefetch(pn + (lane >> 4) * 1024, 0, 1);
    }

    v8f acc = {};
    // K = 0..511 : x_t part
    #pragma unroll
    for (int kc = 0; kc < 16; ++kc) {
        v16bf A  = load8x2(xrow + kc * 32 + ka);
        v16bf Bm = load16 (wrow + kc * 32 + kb);
        acc = __builtin_amdgcn_wmma_f32_16x16x32_bf16(false, A, false, Bm,
                                                      (short)0, acc, false, false);
    }
    // K = 512..1023 : h part
    #pragma unroll
    for (int kc = 0; kc < 16; ++kc) {
        v16bf A  = load8x2(hrow + kc * 32 + ka);
        v16bf Bm = load16 (wrow + EE + kc * 32 + kb);
        acc = __builtin_amdgcn_wmma_f32_16x16x32_bf16(false, A, false, Bm,
                                                      (short)0, acc, false, false);
    }

    const float* bias = dir ? bias_b : bias_f;
    const float bv = bias[nrow];
    const int mbase = mt * 16 + (lane >> 4) * 8;
    #pragma unroll
    for (int r = 0; r < 8; ++r)
        gsm[gc][mbase + r][l15] = acc[r] + bv;

    __syncthreads();

    // Cell update: 2 dirs already split across blocks; this block owns
    // (dir, all 64 batch rows, 16 hidden cols) = 1024 elements / 512 threads.
    #pragma unroll
    for (int e = tid; e < BB * 16; e += 512) {
        const int m  = e >> 4;
        const int jl = e & 15;
        const int j  = jblk * 16 + jl;
        const float ig = gsm[0][m][jl];
        const float fg = gsm[1][m][jl];
        const float gg = gsm[2][m][jl];
        const float og = gsm[3][m][jl];
        const size_t ci = ((size_t)dir * BB + m) * HH + j;
        float c = sigm(fg) * Cbuf[ci] + sigm(ig) * tanhf(gg);
        float h = sigm(og) * tanhf(c);
        Cbuf[ci]   = c;
        Hwrite[ci] = (bf16_t)h;
        hs[((size_t)t * BB + m) * HD2 + dir * HH + j] = (bf16_t)h;
    }
}

// ---------------------------------------------------------------------------
// Kernel 2: feats[b][t][tag] = hs[t][b][:] . W_tag[tag][:] + b_tag[tag]
//   M = L*B = 32768 (2048 tiles), N = 16 (one tile), K = 1024.
// ---------------------------------------------------------------------------
__global__ __launch_bounds__(256)
void tag_gemm_kernel(const bf16_t* __restrict__ hs,
                     const bf16_t* __restrict__ Wtag,   // [T][HD2]
                     const float*  __restrict__ b_tag,
                     float* __restrict__ feats)         // [B][L][T]
{
    const int tid  = threadIdx.x;
    const int lane = tid & 31;
    const int mt   = blockIdx.x * 8 + (tid >> 5);  // 0..2047
    const int l15  = lane & 15;
    const int ka   = (lane >> 4) * 8;
    const int kb   = (lane >> 4) * 8;

    const bf16_t* arow = hs   + ((size_t)mt * 16 + l15) * HD2; // row index = t*B+b
    const bf16_t* wrow = Wtag + (size_t)l15 * HD2;             // tag row (N=16)

    v8f acc = {};
    #pragma unroll
    for (int kc = 0; kc < 32; ++kc) {
        v16bf A  = load8x2(arow + kc * 32 + ka);
        v16bf Bm = load16 (wrow + kc * 32 + kb);
        acc = __builtin_amdgcn_wmma_f32_16x16x32_bf16(false, A, false, Bm,
                                                      (short)0, acc, false, false);
    }
    const float bv = b_tag[l15];
    const int mbase = mt * 16 + (lane >> 4) * 8;
    #pragma unroll
    for (int r = 0; r < 8; ++r) {
        const int m = mbase + r;          // = t*B + b
        const int t = m >> 6;
        const int b = m & (BB - 1);
        feats[(((size_t)b * LSEQ) + t) * TT + l15] = acc[r] + bv;
    }
}

// ---------------------------------------------------------------------------
// Kernel 3: Viterbi per batch row. Writes scores[64] then paths[64*512] (f32).
// ---------------------------------------------------------------------------
__global__ __launch_bounds__(64)
void viterbi_kernel(const float* __restrict__ feats,       // [B][L][T]
                    const float* __restrict__ transitions, // [T][T]
                    unsigned char* __restrict__ bp,        // [B][L][T]
                    float* __restrict__ out)
{
    __shared__ float tr[TT * TT];
    const int tid = threadIdx.x;
    for (int i = tid; i < TT * TT; i += 64) tr[i] = transitions[i];
    __syncthreads();
    if (tid >= BB) return;
    const int b = tid;

    float fv[TT];
    #pragma unroll
    for (int n = 0; n < TT; ++n) fv[n] = NEG_INF;
    fv[START_TAG] = 0.0f;

    const float* fb = feats + (size_t)b * LSEQ * TT;
    unsigned char* bpb = bp + (size_t)b * LSEQ * TT;

    for (int t = 0; t < LSEQ; ++t) {
        float nf[TT];
        #pragma unroll
        for (int n = 0; n < TT; ++n) {
            float best = fv[0] + tr[n * TT];
            int arg = 0;
            #pragma unroll
            for (int p = 1; p < TT; ++p) {
                float sc = fv[p] + tr[n * TT + p];
                if (sc > best) { best = sc; arg = p; }
            }
            nf[n] = best + fb[t * TT + n];
            bpb[t * TT + n] = (unsigned char)arg;
        }
        #pragma unroll
        for (int n = 0; n < TT; ++n) fv[n] = nf[n];
    }
    // terminal transition to STOP
    float best = fv[0] + tr[STOP_TAG * TT];
    int arg = 0;
    #pragma unroll
    for (int p = 1; p < TT; ++p) {
        float sc = fv[p] + tr[STOP_TAG * TT + p];
        if (sc > best) { best = sc; arg = p; }
    }
    out[b] = best;

    float* path = out + BB + (size_t)b * LSEQ;
    int tag = arg;
    for (int t = LSEQ - 1; t >= 0; --t) {
        path[t] = (float)tag;
        tag = bpb[t * TT + tag];
    }
}

// ---------------------------------------------------------------------------
static inline size_t alignup(size_t x) { return (x + 255) & ~(size_t)255; }

extern "C" void kernel_launch(void* const* d_in, const int* in_sizes, int n_in,
                              void* d_out, int out_size, void* d_ws, size_t ws_size,
                              hipStream_t stream) {
    (void)in_sizes; (void)n_in; (void)out_size; (void)ws_size;
    const int*   sentences   = (const int*)  d_in[0];
    const float* embed       = (const float*)d_in[1];
    const float* Wih_f       = (const float*)d_in[2];
    const float* Whh_f       = (const float*)d_in[3];
    const float* b_f         = (const float*)d_in[4];
    const float* Wih_b       = (const float*)d_in[5];
    const float* Whh_b       = (const float*)d_in[6];
    const float* b_b         = (const float*)d_in[7];
    const float* h0          = (const float*)d_in[8];
    const float* c0          = (const float*)d_in[9];
    const float* W_tag       = (const float*)d_in[10];
    const float* b_tag       = (const float*)d_in[11];
    const float* transitions = (const float*)d_in[12];

    char* ws = (char*)d_ws;
    size_t o = 0;
    bf16_t* Xbf  = (bf16_t*)(ws + o); o = alignup(o + (size_t)LSEQ * BB * EE * 2);
    bf16_t* Wcat = (bf16_t*)(ws + o); o = alignup(o + (size_t)2 * GG * HD2 * 2);
    bf16_t* Hbuf = (bf16_t*)(ws + o); o = alignup(o + (size_t)2 * 2 * BB * HH * 2); // [parity][dir][B][H]
    float*  Cbuf = (float*) (ws + o); o = alignup(o + (size_t)2 * BB * HH * 4);
    bf16_t* hs   = (bf16_t*)(ws + o); o = alignup(o + (size_t)LSEQ * BB * HD2 * 2);
    float*  feats= (float*) (ws + o); o = alignup(o + (size_t)BB * LSEQ * TT * 4);
    unsigned char* bp = (unsigned char*)(ws + o); o = alignup(o + (size_t)BB * LSEQ * TT);
    bf16_t* Wtag = (bf16_t*)(ws + o); o = alignup(o + (size_t)TT * HD2 * 2);

    prep_kernel<<<2048, 256, 0, stream>>>(sentences, embed, Wih_f, Whh_f, Wih_b, Whh_b,
                                          h0, c0, W_tag, Xbf, Wcat, Hbuf, Cbuf, Wtag);

    const size_t hstride = (size_t)2 * BB * HH;  // one parity slab (bf16 elements)
    for (int s = 0; s < LSEQ; ++s) {
        const bf16_t* Hread  = Hbuf + (size_t)(s & 1) * hstride;
        bf16_t*       Hwrite = Hbuf + (size_t)((s + 1) & 1) * hstride;
        lstm_step_kernel<<<64, 512, 0, stream>>>(Xbf, Wcat, Hread, Hwrite, Cbuf, hs,
                                                 b_f, b_b, s);
    }

    tag_gemm_kernel<<<256, 256, 0, stream>>>(hs, Wtag, b_tag, feats);
    viterbi_kernel<<<1, 64, 0, stream>>>(feats, transitions, bp, (float*)d_out);
}
